// CNNEncoder_7954279432571
// MI455X (gfx1250) — matrix operand order repaired and checked by reading, not compiled, not run on hardware
//
#include <hip/hip_runtime.h>
#include <hip/hip_bf16.h>
#include <math.h>

// ---- problem constants -----------------------------------------------------
#define B_    16
#define S_    2048
#define N_    3
#define E_    128
#define W_    384        // E*N
#define TWOW_ 768
#define V1_   50001      // V+1
#define L_    12
#define KTAP_ 3
#define LYR_  5
#define SPAD  2050       // S + 2 pad columns
#define SQRT_HALF 0.70710678118654752f

typedef __attribute__((ext_vector_type(16))) _Float16 v16h;
typedef __attribute__((ext_vector_type(8)))  float    v8f;
typedef __attribute__((ext_vector_type(4)))  int      vi4;

// ---- CDNA5 async global->LDS copy (ASYNCcnt) with sync fallback ------------
#if defined(__HIP_DEVICE_COMPILE__) && \
    __has_builtin(__builtin_amdgcn_global_load_async_to_lds_b128)
#define HAVE_ASYNC_LDS 1
// copies 16B per lane: LDS[l + OFF] = MEM[g + OFF]
#define CP16(g, l, OFF)                                                        \
  __builtin_amdgcn_global_load_async_to_lds_b128(                              \
      (__attribute__((address_space(1))) vi4*)(g),                             \
      (__attribute__((address_space(3))) vi4*)(l), (OFF), 0)
#else
#define CP16(g, l, OFF)                                                        \
  do {                                                                         \
    *(uint4*)((char*)(l) + (OFF)) =                                            \
        *(const uint4*)((const char*)(g) + (OFF));                             \
  } while (0)
#endif

__device__ __forceinline__ void wait_async0() {
#if defined(HAVE_ASYNC_LDS)
#if __has_builtin(__builtin_amdgcn_s_wait_asynccnt)
  __builtin_amdgcn_s_wait_asynccnt(0);
#else
  asm volatile("s_wait_asynccnt 0x0" ::: "memory");
#endif
#endif
}

// ---- embedding: e -> Hf32 [b][s][c] (f32) + f16 hi/lo planes [b][s+1][c] ---
__global__ void emb_kernel(const int* __restrict__ x,
                           const int* __restrict__ ngram_ids,
                           const int* __restrict__ ngram_cnt,
                           const float* __restrict__ emb0,
                           const float* __restrict__ tables,
                           float* __restrict__ Hf32,
                           _Float16* __restrict__ phi,
                           _Float16* __restrict__ plo) {
  const int bs = blockIdx.x;            // b*S + s
  const int c  = threadIdx.x;           // 0..127
  const int b  = bs >> 11;
  const int s  = bs & (S_ - 1);
  const int xv = x[bs];
  float out[3];
  if (xv < 4) {
#pragma unroll
    for (int n = 0; n < 3; ++n) out[n] = emb0[xv * W_ + n * E_ + c];
  } else {
#pragma unroll
    for (int n = 0; n < 3; ++n) {
      const int* ids = ngram_ids + ((size_t)bs * 3 + n) * L_;
      float acc = 0.f;
#pragma unroll
      for (int l = 0; l < L_; ++l)
        acc += tables[((size_t)n * V1_ + ids[l]) * E_ + c];   // id==0 rows are 0
      out[n] = acc / (float)ngram_cnt[bs * 3 + n];
    }
  }
#pragma unroll
  for (int n = 0; n < 3; ++n) {
    const float v = out[n];
    Hf32[(size_t)bs * W_ + n * E_ + c] = v;
    const size_t p = ((size_t)b * SPAD + s + 1) * W_ + n * E_ + c;
    const _Float16 hi = (_Float16)v;
    phi[p] = hi;
    plo[p] = (_Float16)(v - (float)hi);
  }
}

// ---- zero the pad columns (s'=0, s'=2049) of both plane ping-pong sets -----
__global__ void pad_zero_kernel(_Float16* p0h, _Float16* p0l,
                                _Float16* p1h, _Float16* p1l) {
  const int i = blockIdx.x * blockDim.x + threadIdx.x;
  if (i >= B_ * W_) return;
  const int b = i / W_, c = i % W_;
  const size_t lo = ((size_t)b * SPAD + 0) * W_ + c;
  const size_t hi = ((size_t)b * SPAD + (SPAD - 1)) * W_ + c;
  p0h[lo] = (_Float16)0.f; p0h[hi] = (_Float16)0.f;
  p0l[lo] = (_Float16)0.f; p0l[hi] = (_Float16)0.f;
  p1h[lo] = (_Float16)0.f; p1h[hi] = (_Float16)0.f;
  p1l[lo] = (_Float16)0.f; p1l[hi] = (_Float16)0.f;
}

// ---- weight-norm reduction (deterministic two-stage) -----------------------
__global__ void norm_partial_kernel(const float* __restrict__ conv_v,
                                    float* __restrict__ partials) {
  const int layer = blockIdx.x / 1152;
  const int blk   = blockIdx.x % 1152;
  const int i     = blk * 256 + threadIdx.x;          // < 294912 (= 768*384)
  const float* v  = conv_v + (size_t)layer * 884736 + (size_t)i * 3;
  __shared__ float sm[3][256];
  sm[0][threadIdx.x] = v[0] * v[0];
  sm[1][threadIdx.x] = v[1] * v[1];
  sm[2][threadIdx.x] = v[2] * v[2];
  __syncthreads();
  for (int st = 128; st > 0; st >>= 1) {
    if (threadIdx.x < st) {
      sm[0][threadIdx.x] += sm[0][threadIdx.x + st];
      sm[1][threadIdx.x] += sm[1][threadIdx.x + st];
      sm[2][threadIdx.x] += sm[2][threadIdx.x + st];
    }
    __syncthreads();
  }
  if (threadIdx.x == 0) {
    float* o = partials + ((size_t)layer * 1152 + blk) * 3;
    o[0] = sm[0][0]; o[1] = sm[1][0]; o[2] = sm[2][0];
  }
}

__global__ void norm_combine_kernel(const float* __restrict__ partials,
                                    float* __restrict__ norm2) {
  const int lk = blockIdx.x;                // layer*3 + k
  const int layer = lk / 3, k = lk % 3;
  float acc = 0.f;
  for (int j = threadIdx.x; j < 1152; j += 256)
    acc += partials[((size_t)layer * 1152 + j) * 3 + k];
  __shared__ float sm[256];
  sm[threadIdx.x] = acc;
  __syncthreads();
  for (int st = 128; st > 0; st >>= 1) {
    if (threadIdx.x < st) sm[threadIdx.x] += sm[threadIdx.x + st];
    __syncthreads();
  }
  if (threadIdx.x == 0) norm2[lk] = sm[0];
}

// ---- scale weights + split to f16 hi/lo planes, layout [layer][k][o][c] ----
__global__ void wscale_kernel(const float* __restrict__ conv_v,
                              const float* __restrict__ conv_g,
                              const float* __restrict__ norm2,
                              _Float16* __restrict__ whi,
                              _Float16* __restrict__ wlo) {
  const size_t idx = (size_t)blockIdx.x * 256 + threadIdx.x;  // < 4423680
  const int lk = (int)(idx / 294912);
  const int oc = (int)(idx % 294912);                          // o*384 + c
  const int layer = lk / 3, k = lk % 3;
  const float scale = conv_g[lk] * rsqrtf(norm2[lk]);
  const float wv = conv_v[(size_t)layer * 884736 + (size_t)oc * 3 + k] * scale;
  const _Float16 hi = (_Float16)wv;
  whi[idx] = hi;
  wlo[idx] = (_Float16)(wv - (float)hi);
}

// ---- f32 transpose [b][s][c] -> [b][c][s] ---------------------------------
__global__ void transpose_kernel(const float* __restrict__ in,
                                 float* __restrict__ out) {
  __shared__ float tile[32][33];
  const int s0 = blockIdx.x * 32, c0 = blockIdx.y * 32, b = blockIdx.z;
#pragma unroll
  for (int i = 0; i < 4; ++i) {
    const int s = s0 + threadIdx.y + i * 8;
    tile[threadIdx.y + i * 8][threadIdx.x] =
        in[((size_t)b * S_ + s) * W_ + c0 + threadIdx.x];
  }
  __syncthreads();
#pragma unroll
  for (int i = 0; i < 4; ++i) {
    const int c = c0 + threadIdx.y + i * 8;
    out[((size_t)b * W_ + c) * S_ + s0 + threadIdx.x] =
        tile[threadIdx.x][threadIdx.y + i * 8];
  }
}

// ---- fused conv(GEMM, split-f16 x3 WMMA) + bias + GLU + residual ----------
// 256 thr (8 waves). Paired 128-row a-half (o = mb*128+m) and b-half (o+384)
// over a 128-wide s-tile; GLU pairs live in the same lane/accumulator slot.
// LDS double-buffered; tile fills go through the CDNA5 async global->LDS path
// (ASYNCcnt) and overlap with the WMMA stage; one barrier per K-slab.
__global__ __launch_bounds__(256, 1) void conv_gemm_kernel(
    const _Float16* __restrict__ whi,     // layer base: [3][768][384]
    const _Float16* __restrict__ wlo,
    const _Float16* __restrict__ phi_in,  // [16][2050][384]
    const _Float16* __restrict__ plo_in,
    const float* __restrict__ bias,       // [768] (this layer)
    float* __restrict__ Hf32,             // [16][2048][384] in/out (residual)
    _Float16* __restrict__ phi_out,       // next-layer planes
    _Float16* __restrict__ plo_out,
    int write_planes) {
  const int mb = blockIdx.x;   // 0..2  (row block of the a-half)
  const int st = blockIdx.y;   // 0..15 (s tile)
  const int b  = blockIdx.z;   // 0..15
  const int tid = threadIdx.x;
  const int w  = tid >> 5;     // wave 0..7
  const int l  = tid & 31;
  const int lm = l & 15;
  const int lh = l >> 4;

  // double-buffered, rows padded to 40 f16 (80B) -> conflict-free b128 frags
  __shared__ __align__(16) _Float16 Ah[2][256 * 40];
  __shared__ __align__(16) _Float16 Al[2][256 * 40];
  __shared__ __align__(16) _Float16 Bh[2][128 * 40];
  __shared__ __align__(16) _Float16 Bl[2][128 * 40];

  v8f acc[2][8];
  const v8f vzero = {0.f, 0.f, 0.f, 0.f, 0.f, 0.f, 0.f, 0.f};
#pragma unroll
  for (int h = 0; h < 2; ++h)
#pragma unroll
    for (int j = 0; j < 8; ++j) acc[h][j] = vzero;

  const int s_base = st * 128;
  const int arow_g = (tid < 128) ? (mb * 128 + tid)
                                 : (384 + mb * 128 + (tid - 128));
  const int bn = tid >> 1;          // B row (s position within tile)
  const int bo = (tid & 1) * 16;    // f16 offset within the 32-wide K slab

  union F { uint4 q[2]; v16h v; };

  // issue tile fills for flattened iteration `it` into buffer `p`
  auto fill = [&](int it, int p) {
    const int k  = it / 12;          // conv tap
    const int cb = (it % 12) * 32;   // K slab base (input channel)
    // A: 256 rows x 32 K, hi+lo; thread t owns one 64B row
    {
      const _Float16* gh = whi + (size_t)k * 294912 + (size_t)arow_g * W_ + cb;
      const _Float16* gl = wlo + (size_t)k * 294912 + (size_t)arow_g * W_ + cb;
      _Float16* dh = &Ah[p][tid * 40];
      _Float16* dl = &Al[p][tid * 40];
      CP16(gh, dh, 0);  CP16(gh, dh, 16); CP16(gh, dh, 32); CP16(gh, dh, 48);
      CP16(gl, dl, 0);  CP16(gl, dl, 16); CP16(gl, dl, 32); CP16(gl, dl, 48);
    }
    // B: 128 s-rows x 32 K, hi+lo; 2 threads per row (32B each)
    {
      const size_t brow =
          ((size_t)b * SPAD + s_base + k + bn) * W_ + cb + bo;
      const _Float16* gh = phi_in + brow;
      const _Float16* gl = plo_in + brow;
      _Float16* dh = &Bh[p][bn * 40 + bo];
      _Float16* dl = &Bl[p][bn * 40 + bo];
      CP16(gh, dh, 0); CP16(gh, dh, 16);
      CP16(gl, dl, 0); CP16(gl, dl, 16);
    }
  };

  fill(0, 0);
#pragma unroll 1
  for (int it = 0; it < KTAP_ * 12; ++it) {
    const int p = it & 1;
    wait_async0();        // my fills for buffer p have landed
    __syncthreads();      // everyone's fills visible; prev compute done
    if (it + 1 < KTAP_ * 12) fill(it + 1, p ^ 1);  // overlap with WMMA below

    // --- A fragments (ISA layout: K 0-7/8-15 interleaved per lane-half) ---
    v16h ahi[2], alo[2];
#pragma unroll
    for (int h = 0; h < 2; ++h) {
      const int row = h * 128 + w * 16 + lm;
      F fa; fa.q[0] = *(const uint4*)(&Ah[p][row * 40 + lh * 8]);
            fa.q[1] = *(const uint4*)(&Ah[p][row * 40 + 16 + lh * 8]);
      ahi[h] = fa.v;
      F fl; fl.q[0] = *(const uint4*)(&Al[p][row * 40 + lh * 8]);
            fl.q[1] = *(const uint4*)(&Al[p][row * 40 + 16 + lh * 8]);
      alo[h] = fl.v;
    }
    // --- B fragments (ISA layout: K 0-7/8-15 interleaved per lane-half) + 3x WMMA ---
#pragma unroll
    for (int j = 0; j < 8; ++j) {
      const int col = j * 16 + lm;
      F fb; fb.q[0] = *(const uint4*)(&Bh[p][col * 40 + lh * 8]);
            fb.q[1] = *(const uint4*)(&Bh[p][col * 40 + 16 + lh * 8]);
      const v16h bhi = fb.v;
      F fl; fl.q[0] = *(const uint4*)(&Bl[p][col * 40 + lh * 8]);
            fl.q[1] = *(const uint4*)(&Bl[p][col * 40 + 16 + lh * 8]);
      const v16h blo = fl.v;
#pragma unroll
      for (int h = 0; h < 2; ++h) {
        acc[h][j] = __builtin_amdgcn_wmma_f32_16x16x32_f16(
            false, ahi[h], false, bhi, (short)0, acc[h][j], false, false);
        acc[h][j] = __builtin_amdgcn_wmma_f32_16x16x32_f16(
            false, ahi[h], false, blo, (short)0, acc[h][j], false, false);
        acc[h][j] = __builtin_amdgcn_wmma_f32_16x16x32_f16(
            false, alo[h], false, bhi, (short)0, acc[h][j], false, false);
      }
    }
  }

  // --- epilogue: bias, GLU, residual, write f32 + next-layer f16 planes ---
  const int obase = mb * 128 + w * 16 + 8 * lh;   // 8 consecutive channels
  float ba[8], bb[8];
#pragma unroll
  for (int r = 0; r < 8; ++r) {
    ba[r] = bias[obase + r];
    bb[r] = bias[W_ + obase + r];
  }
#pragma unroll
  for (int j = 0; j < 8; ++j) {
    const int s = s_base + j * 16 + lm;
    float* hp = Hf32 + ((size_t)b * S_ + s) * W_ + obase;
    const size_t pofs = ((size_t)b * SPAD + s + 1) * W_ + obase;
#pragma unroll
    for (int r = 0; r < 8; ++r) {
      const float a  = acc[0][j][r] + ba[r];
      const float g  = acc[1][j][r] + bb[r];
      const float y  = a / (1.f + expf(-g));       // a * sigmoid(g)
      const float hn = (y + hp[r]) * SQRT_HALF;
      hp[r] = hn;
      if (write_planes) {
        const _Float16 h16 = (_Float16)hn;
        phi_out[pofs + r] = h16;
        plo_out[pofs + r] = (_Float16)(hn - (float)h16);
      }
    }
  }
}

// ---------------------------------------------------------------------------
extern "C" void kernel_launch(void* const* d_in, const int* in_sizes, int n_in,
                              void* d_out, int out_size, void* d_ws,
                              size_t ws_size, hipStream_t stream) {
  (void)in_sizes; (void)n_in; (void)out_size; (void)ws_size;
  const int*   x         = (const int*)d_in[0];
  const int*   ngram_ids = (const int*)d_in[1];
  const int*   ngram_cnt = (const int*)d_in[2];
  const float* emb0      = (const float*)d_in[3];
  const float* tables    = (const float*)d_in[4];
  const float* conv_v    = (const float*)d_in[5];
  const float* conv_g    = (const float*)d_in[6];
  const float* conv_b    = (const float*)d_in[7];

  // workspace layout (bytes)
  char* ws = (char*)d_ws;
  const size_t HF32_SZ  = (size_t)B_ * S_ * W_ * 4;      // 50,331,648
  const size_t PLANE_SZ = (size_t)B_ * SPAD * W_ * 2;    // 25,190,400
  const size_t WPL_SZ   = (size_t)LYR_ * 3 * TWOW_ * W_ * 2; // 8,847,360
  float*    Hf32 = (float*)(ws);
  _Float16* P0H  = (_Float16*)(ws + HF32_SZ);
  _Float16* P0L  = (_Float16*)(ws + HF32_SZ + PLANE_SZ);
  _Float16* P1H  = (_Float16*)(ws + HF32_SZ + 2 * PLANE_SZ);
  _Float16* P1L  = (_Float16*)(ws + HF32_SZ + 3 * PLANE_SZ);
  _Float16* WHI  = (_Float16*)(ws + HF32_SZ + 4 * PLANE_SZ);
  _Float16* WLO  = (_Float16*)(ws + HF32_SZ + 4 * PLANE_SZ + WPL_SZ);
  float*    PART = (float*)(ws + HF32_SZ + 4 * PLANE_SZ + 2 * WPL_SZ);
  float*    NRM2 = (float*)(ws + HF32_SZ + 4 * PLANE_SZ + 2 * WPL_SZ + 69120);

  float* h_out = (float*)d_out;                        // (B, W, S)
  float* e_out = (float*)d_out + (size_t)B_ * W_ * S_; // (B, W, S)

  // 1) zero pad columns of both plane sets
  pad_zero_kernel<<<(B_ * W_ + 255) / 256, 256, 0, stream>>>(P0H, P0L, P1H, P1L);
  // 2) embeddings -> Hf32 + plane set 0
  emb_kernel<<<B_ * S_, 128, 0, stream>>>(x, ngram_ids, ngram_cnt, emb0, tables,
                                          Hf32, P0H, P0L);
  // 3) e output (transpose before conv layers overwrite Hf32 in place)
  transpose_kernel<<<dim3(S_ / 32, W_ / 32, B_), dim3(32, 8), 0, stream>>>(
      Hf32, e_out);
  // 4) weight norms (deterministic two-stage reduce) + scaled f16 weights
  norm_partial_kernel<<<1152 * LYR_, 256, 0, stream>>>(conv_v, PART);
  norm_combine_kernel<<<LYR_ * 3, 256, 0, stream>>>(PART, NRM2);
  wscale_kernel<<<(LYR_ * 3 * TWOW_ * W_) / 256, 256, 0, stream>>>(
      conv_v, conv_g, NRM2, WHI, WLO);
  // 5) conv stack: ping-pong f16 planes, f32 h updated in place
  for (int layer = 0; layer < LYR_; ++layer) {
    _Float16* pih = (layer & 1) ? P1H : P0H;
    _Float16* pil = (layer & 1) ? P1L : P0L;
    _Float16* poh = (layer & 1) ? P0H : P1H;
    _Float16* pol = (layer & 1) ? P0L : P1L;
    conv_gemm_kernel<<<dim3(3, 16, 16), 256, 0, stream>>>(
        WHI + (size_t)layer * 884736, WLO + (size_t)layer * 884736,
        pih, pil, conv_b + layer * TWOW_, Hf32, poh, pol,
        (layer < LYR_ - 1) ? 1 : 0);
  }
  // 6) h output
  transpose_kernel<<<dim3(S_ / 32, W_ / 32, B_), dim3(32, 8), 0, stream>>>(
      Hf32, h_out);
}
